// ReplicatedLinear_76690936037442
// MI455X (gfx1250) — matrix operand order repaired and checked by reading, compiled, not run on hardware
//
#include <hip/hip_runtime.h>

// ---------------------------------------------------------------------------
// Types
// ---------------------------------------------------------------------------
typedef __attribute__((ext_vector_type(16))) __bf16        v16bf;
typedef __attribute__((ext_vector_type(8)))  float          v8f;
typedef __attribute__((ext_vector_type(4)))  float          f32x4;
typedef __attribute__((ext_vector_type(4)))  unsigned int   u32x4;
typedef __attribute__((ext_vector_type(4)))  unsigned short u16x4;

// GCC-style vector to exactly match the async builtin's parameter type
typedef int v4i __attribute__((vector_size(4 * sizeof(int))));
#define AS1 __attribute__((address_space(1)))
#define AS3 __attribute__((address_space(3)))

__device__ __forceinline__ unsigned short f32_to_bf16(float f) {
  unsigned int u = __float_as_uint(f);
  u += 0x7FFFu + ((u >> 16) & 1u);        // round-to-nearest-even
  return (unsigned short)(u >> 16);
}

// ---------------------------------------------------------------------------
// CDNA5 async global->LDS path (gated; sync fallback keeps compile green)
// ---------------------------------------------------------------------------
#if defined(__has_builtin)
#  if __has_builtin(__builtin_amdgcn_global_load_async_to_lds_b128)
#    define HAVE_ASYNC_LDS 1
#  endif
#endif
#ifndef HAVE_ASYNC_LDS
#  define HAVE_ASYNC_LDS 0
#endif

__device__ __forceinline__ void wait_async_zero() {
#if HAVE_ASYNC_LDS
#  if __has_builtin(__builtin_amdgcn_s_wait_asynccnt)
  __builtin_amdgcn_s_wait_asynccnt(0);
#  else
  asm volatile("s_wait_asynccnt 0x0" ::: "memory");
#  endif
#endif
}

// ---------------------------------------------------------------------------
// Problem constants (fixed by the reference)
// ---------------------------------------------------------------------------
#define M_TOT 8192      // B*S
#define K_TOT 4096      // D_IN
#define N_TOT 4096      // D_OUT
#define LORA_R 16
#define SCALING 2.0f    // 32/16

#define BM 256
#define BN 128
#define BK 64
#define KTILES (K_TOT / BK)
#define LDA 72          // padded row stride in halves (144 B, 16B aligned)
#define LDB 72

// ---------------------------------------------------------------------------
// Kernel 1: fuse dequant + LoRA into one bf16 weight:
//   W_c[o,i] = (float)q[o,i]*scales[o] + SCALING * sum_r lB[o,r]*lA[r,i]
// ---------------------------------------------------------------------------
__global__ __launch_bounds__(256)
void combine_kernel(const signed char* __restrict__ q,
                    const float* __restrict__ scales,
                    const float* __restrict__ lA,
                    const float* __restrict__ lB,
                    unsigned short* __restrict__ wc) {
  const int o  = blockIdx.y;
  const int i0 = (blockIdx.x * 256 + threadIdx.x) * 4;
  const float s = scales[o];

  f32x4 acc = (f32x4){0.f, 0.f, 0.f, 0.f};
#pragma unroll
  for (int r = 0; r < LORA_R; ++r) {
    const float br = lB[o * LORA_R + r];
    f32x4 a = *(const f32x4*)(lA + r * K_TOT + i0);
    acc += a * br;
  }

  const int packed = *(const int*)(q + (size_t)o * K_TOT + i0);
  u16x4 h;
  h.x = f32_to_bf16((float)(signed char)((packed      ) & 0xff) * s + SCALING * acc.x);
  h.y = f32_to_bf16((float)(signed char)((packed >>  8) & 0xff) * s + SCALING * acc.y);
  h.z = f32_to_bf16((float)(signed char)((packed >> 16) & 0xff) * s + SCALING * acc.z);
  h.w = f32_to_bf16((float)(signed char)((packed >> 24) & 0xff) * s + SCALING * acc.w);
  *(u16x4*)(wc + (size_t)o * K_TOT + i0) = h;
}

// ---------------------------------------------------------------------------
// GEMM helpers
// ---------------------------------------------------------------------------
__device__ __forceinline__ void load_a_tile(f32x4 areg[16], const float* __restrict__ x,
                                            int mBase, int kt, int tid) {
#pragma unroll
  for (int j = 0; j < 16; ++j) {
    const int idx = tid + j * 256;
    const int row = idx >> 4;
    const int c4  = idx & 15;
    areg[j] = *(const f32x4*)(x + (size_t)(mBase + row) * K_TOT + kt + c4 * 4);
  }
}

__device__ __forceinline__ void store_a_tile(unsigned short* As, const f32x4 areg[16], int tid) {
#pragma unroll
  for (int j = 0; j < 16; ++j) {
    const int idx = tid + j * 256;
    const int row = idx >> 4;
    const int c4  = idx & 15;
    u16x4 h;
    h.x = f32_to_bf16(areg[j].x);
    h.y = f32_to_bf16(areg[j].y);
    h.z = f32_to_bf16(areg[j].z);
    h.w = f32_to_bf16(areg[j].w);
    *(u16x4*)(&As[row * LDA + c4 * 4]) = h;
  }
}

__device__ __forceinline__ void load_w_tile(unsigned short* Wl, const unsigned short* __restrict__ wc,
                                            int nBase, int kt, int tid) {
#pragma unroll
  for (int j = 0; j < 4; ++j) {
    const int idx = tid + j * 256;
    const int row = idx >> 3;
    const int c8  = idx & 7;
    const unsigned short* g = wc + (size_t)(nBase + row) * K_TOT + kt + c8 * 8;
    unsigned short* l = Wl + row * LDB + c8 * 8;
#if HAVE_ASYNC_LDS
    __builtin_amdgcn_global_load_async_to_lds_b128(
        (AS1 v4i*)(unsigned long long)(uintptr_t)g,
        (AS3 v4i*)(unsigned int)(uintptr_t)l,
        0, 0);
#else
    *(u32x4*)l = *(const u32x4*)g;
#endif
  }
}

// ---------------------------------------------------------------------------
// Kernel 2: out[M,N] = bf16(x)[M,K] @ W_c[N,K]^T + bias  via v_wmma bf16
// block 256 threads (8 waves); block tile 256x128x64; wave tile 64x64.
// Double-buffered LDS, one barrier per K-tile; A pipelined through registers,
// W streamed with global_load_async_to_lds_b128 (ASYNCcnt).
// ---------------------------------------------------------------------------
__global__ __launch_bounds__(256)
void gemm_kernel(const float* __restrict__ x,
                 const unsigned short* __restrict__ wc,
                 const float* __restrict__ bias,
                 float* __restrict__ out) {
  __shared__ __align__(16) unsigned short As[2][BM * LDA];  // 2 x 36,864 B
  __shared__ __align__(16) unsigned short Ws[2][BN * LDB];  // 2 x 18,432 B

  const int tid   = threadIdx.x;
  const int lane  = tid & 31;
  const int wave  = tid >> 5;
  const int waveM = wave >> 1;                 // 0..3
  const int waveN = wave & 1;                  // 0..1
  const int mBase = blockIdx.y * BM;
  const int nBase = blockIdx.x * BN;

  union AB { v16bf v; u32x4 q[2]; };

  v8f acc[4][4];
  const v8f vzero = (v8f){0.f,0.f,0.f,0.f,0.f,0.f,0.f,0.f};
#pragma unroll
  for (int mi = 0; mi < 4; ++mi)
#pragma unroll
    for (int ni = 0; ni < 4; ++ni)
      acc[mi][ni] = vzero;

  const int lrow = lane & 15;
  const int aoff = (lane >> 4) * 8;    // A-frag K sub-offset (halves)
  const int boff = (lane >> 4) * 8;    // B-frag K sub-offset (halves)

  // ---- prologue: tile 0 in flight ----
  f32x4 areg[16];
  load_a_tile(areg, x, mBase, 0, tid);
  load_w_tile(&Ws[0][0], wc, nBase, 0, tid);

  for (int t = 0; t < KTILES; ++t) {
    const int buf = t & 1;

    // stage A(t) into LDS; prefetch A(t+1) into registers
    store_a_tile(&As[buf][0], areg, tid);
    if (t + 1 < KTILES)
      load_a_tile(areg, x, mBase, (t + 1) * BK, tid);

    wait_async_zero();           // W(t) landed in Ws[buf]
    __syncthreads();             // publish As[buf] (DScnt drained by compiler) + Ws[buf]

    // issue W(t+1): Ws[buf^1] was last read in compute(t-1), done for all waves
    if (t + 1 < KTILES)
      load_w_tile(&Ws[buf ^ 1][0], wc, nBase, (t + 1) * BK, tid);

    // ---- compute on As[buf], Ws[buf] ----
#pragma unroll
    for (int ks = 0; ks < BK; ks += 32) {
      AB a[4], b[4];
#pragma unroll
      for (int mi = 0; mi < 4; ++mi) {
        const unsigned short* p = &As[buf][(waveM * 64 + mi * 16 + lrow) * LDA + ks];
        a[mi].q[0] = *(const u32x4*)(p + aoff);        // K {0..7}/{8..15}
        a[mi].q[1] = *(const u32x4*)(p + 16 + aoff);   // K {16..23}/{24..31}
      }
#pragma unroll
      for (int ni = 0; ni < 4; ++ni) {
        const unsigned short* p = &Ws[buf][(waveN * 64 + ni * 16 + lrow) * LDB + ks + boff];
        b[ni].q[0] = *(const u32x4*)(p);               // K {0..7}/{8..15}
        b[ni].q[1] = *(const u32x4*)(p + 16);          // K {16..23}/{24..31}
      }
#pragma unroll
      for (int mi = 0; mi < 4; ++mi)
#pragma unroll
        for (int ni = 0; ni < 4; ++ni)
          acc[mi][ni] = __builtin_amdgcn_wmma_f32_16x16x32_bf16(
              false, a[mi].v, false, b[ni].v,
              (short)0, acc[mi][ni], false, false);
    }
  }

  // ---- epilogue: add bias, store fp32 ----
  const int rOff = (lane >> 4) * 8;   // C/D layout: lanes 16-31 hold M+8
#pragma unroll
  for (int ni = 0; ni < 4; ++ni) {
    const int col = nBase + waveN * 64 + ni * 16 + lrow;
    const float bv = bias[col];
#pragma unroll
    for (int mi = 0; mi < 4; ++mi) {
      const int m0 = mBase + waveM * 64 + mi * 16 + rOff;
      v8f c = acc[mi][ni];
#pragma unroll
      for (int v = 0; v < 8; ++v)
        out[(size_t)(m0 + v) * N_TOT + col] = c[v] + bv;
    }
  }
}

// ---------------------------------------------------------------------------
// Launch
// ---------------------------------------------------------------------------
extern "C" void kernel_launch(void* const* d_in, const int* in_sizes, int n_in,
                              void* d_out, int out_size, void* d_ws, size_t ws_size,
                              hipStream_t stream) {
  const float*       x      = (const float*)d_in[0];
  const signed char* q      = (const signed char*)d_in[1];
  const float*       scales = (const float*)d_in[2];
  const float*       bias   = (const float*)d_in[3];
  const float*       lA     = (const float*)d_in[4];
  const float*       lB     = (const float*)d_in[5];
  float*             out    = (float*)d_out;

  // workspace: combined bf16 weight [N_TOT, K_TOT] = 32 MiB
  unsigned short* wc = (unsigned short*)d_ws;

  combine_kernel<<<dim3(K_TOT / (256 * 4), N_TOT), 256, 0, stream>>>(q, scales, lA, lB, wc);
  gemm_kernel<<<dim3(N_TOT / BN, M_TOT / BM), 256, 0, stream>>>(x, wc, bias, out);
}
